// TransducerBeamSearcherforFairseq_80779744903215
// MI455X (gfx1250) — compile-verified, matrix operand order checked by reading, not run on hardware
//
#include <hip/hip_runtime.h>
#include <hip/hip_bf16.h>
#include <cstdint>

#define BDIM 1024
#define BB 32
#define TT 1000
#define DD 640
#define VV 1024

#if defined(__HIP_DEVICE_COMPILE__)
#if !__has_builtin(__builtin_amdgcn_wmma_f32_16x16x32_bf16)
#error "wmma bf16 builtin missing on this toolchain (device pass)"
#endif
#endif

typedef __attribute__((ext_vector_type(16))) __bf16 v16bf;
typedef __attribute__((ext_vector_type(8)))  __bf16 v8bf;
typedef __attribute__((ext_vector_type(4)))  __bf16 v4bf;
typedef __attribute__((ext_vector_type(8)))  float  v8f;

__device__ __forceinline__ v8f zero8() {
  v8f z;
#pragma unroll
  for (int i = 0; i < 8; ++i) z[i] = 0.0f;
  return z;
}

__device__ __forceinline__ v8f wmma_bf16(v16bf a, v16bf b, v8f c) {
  // D = A(16x32 bf16) x B(32x16 bf16) + C(16x16 f32)
  return __builtin_amdgcn_wmma_f32_16x16x32_bf16(false, a, false, b, (short)0, c,
                                                 false, false);
}

// B fragment from a row-major f32 weight row (global, L2-resident):
// K pattern per ISA 7.12.2: half0 -> k0+{0..7,16..23}, half1 -> k0+{8..15,24..31}
__device__ __forceinline__ v16bf mk_frag_f32(const float* __restrict__ p, int half) {
  v16bf f;
  p += half * 8;
#pragma unroll
  for (int i = 0; i < 8; ++i) {
    f[i]     = (__bf16)p[i];
    f[i + 8] = (__bf16)p[i + 16];
  }
  return f;
}

// A fragment from a bf16 LDS row: two aligned 16B loads, no conversion VALU.
__device__ __forceinline__ v16bf ld_frag_bf(const __bf16* p, int half) {
  p += half * 8;
  v8bf lo = *(const v8bf*)(p);
  v8bf hi = *(const v8bf*)(p + 16);
  v16bf f;
#pragma unroll
  for (int i = 0; i < 8; ++i) {
    f[i]     = lo[i];
    f[i + 8] = hi[i];
  }
  return f;
}

__device__ __forceinline__ float sig_fast(float x) {
  return 1.0f / (1.0f + __expf(-x));
}
__device__ __forceinline__ float tanh_fast(float x) {
  float e = __expf(-2.0f * fabsf(x));           // branchless, v_exp_f32
  float t = 1.0f - 2.0f * e / (1.0f + e);
  return copysignf(t, x);
}

// order-preserving float -> u32 key (monotonic)
__device__ __forceinline__ unsigned int fkey(float x) {
  unsigned int u = __float_as_uint(x);
  return (u & 0x80000000u) ? ~u : (u | 0x80000000u);
}

// One gate's GEMM: x_tile @ Wih_row^T + h_tile @ Whh_row^T (K=640).
// Two independent accumulators halve the WMMA RAW chain; summed at the end.
__device__ __forceinline__ v8f gate_gemm(const __bf16* __restrict__ xrow,
                                         const __bf16* __restrict__ hrow,
                                         const float* __restrict__ wih_row,
                                         const float* __restrict__ whh_row,
                                         bool init, int half) {
  v8f accx = zero8();
  v8f acch = zero8();
  for (int k0 = 0; k0 < DD; k0 += 32) {
    __builtin_prefetch(wih_row + k0 + 32, 0, 3);
    accx = wmma_bf16(ld_frag_bf(xrow + k0, half), mk_frag_f32(wih_row + k0, half), accx);
    if (!init)  // h0 == 0 at init: skip hidden GEMM entirely
      acch = wmma_bf16(ld_frag_bf(hrow + k0, half), mk_frag_f32(whh_row + k0, half), acch);
  }
#pragma unroll
  for (int v = 0; v < 8; ++v) accx[v] += acch[v];
  return accx;
}

// LSTM phase, gate-sequential (f -> i -> g -> o) to cap live v8f registers.
// Unit u = wave+32q: row-tile r=u&1, col group jg=u>>1. c stays in registers.
__device__ __forceinline__ void lstm_phase(
    const __bf16* __restrict__ xe, const __bf16* __restrict__ hinb,
    __bf16* __restrict__ houtb, const float* __restrict__ Wih,
    const float* __restrict__ Whh, const float* __restrict__ bl, v8f* creg,
    const int* __restrict__ nbS, bool init, int wave, int l16, int half) {
  for (int q = 0; q < 3; ++q) {
    int u = wave + 32 * q;
    if (u >= 80) break;  // wave-uniform: EXEC stays all-ones around WMMA
    int r = u & 1, jg = u >> 1;
    int j = jg * 16 + l16;
    const __bf16* xr = xe + (r * 16 + l16) * DD;
    const __bf16* hr = hinb + (r * 16 + l16) * DD;

    // forget gate
    v8f sf = gate_gemm(xr, hr, Wih + (size_t)(1 * DD + j) * DD,
                       Whh + (size_t)(1 * DD + j) * DD, init, half);
    float bfv = bl[1 * DD + j];
#pragma unroll
    for (int v = 0; v < 8; ++v) sf[v] = sig_fast(sf[v] + bfv);

    // input gate
    v8f si = gate_gemm(xr, hr, Wih + (size_t)(0 * DD + j) * DD,
                       Whh + (size_t)(0 * DD + j) * DD, init, half);
    float bi = bl[0 * DD + j];
#pragma unroll
    for (int v = 0; v < 8; ++v) si[v] = sig_fast(si[v] + bi);

    // cell candidate -> new cell (folds in sf, si; they die here)
    v8f cn = gate_gemm(xr, hr, Wih + (size_t)(2 * DD + j) * DD,
                       Whh + (size_t)(2 * DD + j) * DD, init, half);
    float bg = bl[2 * DD + j];
#pragma unroll
    for (int v = 0; v < 8; ++v)
      cn[v] = sf[v] * creg[q][v] + si[v] * tanh_fast(cn[v] + bg);

    // output gate -> h, masked merge
    v8f ho = gate_gemm(xr, hr, Wih + (size_t)(3 * DD + j) * DD,
                       Whh + (size_t)(3 * DD + j) * DD, init, half);
    float bov = bl[3 * DD + j];
#pragma unroll
    for (int v = 0; v < 8; ++v) {
      int m = r * 16 + v + 8 * half;  // C/D layout: m = v + 8*half
      int nb = init ? 1 : nbS[m];
      float hnv = sig_fast(ho[v] + bov) * tanh_fast(cn[v]);
      float hold = init ? 0.0f : (float)hinb[m * DD + j];
      creg[q][v] = nb ? cn[v] : creg[q][v];
      houtb[m * DD + j] = (__bf16)(nb ? hnv : hold);
    }
  }
}

__global__ __launch_bounds__(BDIM) void rnnt_decode(
    const float* __restrict__ tn, const float* __restrict__ E,
    const float* __restrict__ Wih, const float* __restrict__ Whh,
    const float* __restrict__ bl, const float* __restrict__ Wtn,
    const float* __restrict__ Wpn, const float* __restrict__ bj,
    const float* __restrict__ Wout, const float* __restrict__ bo,
    float* __restrict__ out) {
  extern __shared__ char smem[];
  __bf16* hbA = (__bf16*)smem;                    // h buffer A [32][640] bf16
  __bf16* hbB = (__bf16*)(smem + 40960);          // h buffer B
  __bf16* zxb = (__bf16*)(smem + 81920);          // shared: z / x-embedding
  __bf16* tnb = (__bf16*)(smem + 122880);         // staged tn_t slice
  char* sc = smem + 163840;
  unsigned long long* pmax = (unsigned long long*)sc;  // packed (key|~idx)/row
  float* sumexp = (float*)(sc + 256);
  float* score  = (float*)(sc + 384);
  int* tokS     = (int*)(sc + 512);
  int* lastS    = (int*)(sc + 640);
  int* nbS      = (int*)(sc + 768);

  const int tid = threadIdx.x;
  const int wave = tid >> 5;
  const int lane = tid & 31;
  const int l16 = lane & 15;
  const int half = lane >> 4;

  v8f creg[3];
#pragma unroll
  for (int q = 0; q < 3; ++q) creg[q] = zero8();

  if (tid < 32) { score[tid] = 0.0f; lastS[tid] = 0; }

  // ---- init: x0 = E[BLANK] broadcast to all batch rows (bf16 stage) ----
  for (int i = tid; i < BB * (DD / 4); i += BDIM) {
    int c4 = i % (DD / 4);
    float4 fv = ((const float4*)E)[c4];
    v4bf w; w[0] = (__bf16)fv.x; w[1] = (__bf16)fv.y; w[2] = (__bf16)fv.z; w[3] = (__bf16)fv.w;
    *(v4bf*)(zxb + (size_t)i * 4) = w;
  }
  __syncthreads();
  lstm_phase(zxb, hbA, hbA, Wih, Whh, bl, creg, nbS, true, wave, l16, half);
  __syncthreads();

  for (int t = 0; t < TT; ++t) {
    const __bf16* hc = (t & 1) ? hbB : hbA;
    __bf16* hnx = (t & 1) ? hbA : hbB;
    if (tid < 32) { pmax[tid] = 0ull; sumexp[tid] = 0.0f; }

    // ---- phase 0: stage tn_t slice into bf16 LDS (one conversion/element) --
    for (int i = tid; i < BB * (DD / 4); i += BDIM) {
      int row = i / (DD / 4);
      int c4 = i % (DD / 4);
      float4 fv = *(const float4*)(tn + ((size_t)row * TT + t) * DD + c4 * 4);
      v4bf w; w[0] = (__bf16)fv.x; w[1] = (__bf16)fv.y; w[2] = (__bf16)fv.z; w[3] = (__bf16)fv.w;
      *(v4bf*)(tnb + (size_t)i * 4) = w;
    }
    __syncthreads();

    // ---- phase 1: z = tanh(tn_t@W_tn^T + h@W_pn^T + b_joint), 40 units x 2 tiles --
    for (int q = 0; q < 2; ++q) {
      int u = wave + 32 * q;
      if (u >= 40) break;
      int j = u * 16 + l16;
      v8f acc[2]; acc[0] = zero8(); acc[1] = zero8();
      const float* wt = Wtn + (size_t)j * DD;
      const float* wp = Wpn + (size_t)j * DD;
      for (int k0 = 0; k0 < DD; k0 += 32) {
        __builtin_prefetch(wt + k0 + 32, 0, 3);
        v16bf b = mk_frag_f32(wt + k0, half);
        acc[0] = wmma_bf16(ld_frag_bf(tnb + l16 * DD + k0, half), b, acc[0]);
        acc[1] = wmma_bf16(ld_frag_bf(tnb + (16 + l16) * DD + k0, half), b, acc[1]);
        b = mk_frag_f32(wp + k0, half);
        acc[0] = wmma_bf16(ld_frag_bf(hc + l16 * DD + k0, half), b, acc[0]);
        acc[1] = wmma_bf16(ld_frag_bf(hc + (16 + l16) * DD + k0, half), b, acc[1]);
      }
      float bjv = bj[j];
#pragma unroll
      for (int r = 0; r < 2; ++r)
#pragma unroll
        for (int v = 0; v < 8; ++v) {
          int m = r * 16 + v + 8 * half;
          zxb[m * DD + j] = (__bf16)tanh_fast(acc[r][v] + bjv);
        }
    }
    __syncthreads();

    // ---- phase 2a: logits + global argmax via packed ds_max_u64 ----
    v8f st[2][2];
    for (int q = 0; q < 2; ++q) {
      int vg = wave + 32 * q;               // 64 column groups
      int vcol = vg * 16 + l16;
      v8f acc[2]; acc[0] = zero8(); acc[1] = zero8();
      const float* wo = Wout + (size_t)vcol * DD;
      for (int k0 = 0; k0 < DD; k0 += 32) {
        v16bf b = mk_frag_f32(wo + k0, half);
        acc[0] = wmma_bf16(ld_frag_bf(zxb + l16 * DD + k0, half), b, acc[0]);
        acc[1] = wmma_bf16(ld_frag_bf(zxb + (16 + l16) * DD + k0, half), b, acc[1]);
      }
      float bv = bo[vcol];
#pragma unroll
      for (int r = 0; r < 2; ++r) {
#pragma unroll
        for (int v = 0; v < 8; ++v) acc[r][v] += bv;
        st[q][r] = acc[r];
#pragma unroll
        for (int v = 0; v < 8; ++v) {
          int m = r * 16 + v + 8 * half;
          unsigned long long pk =
              ((unsigned long long)fkey(acc[r][v]) << 32) |
              (unsigned long long)(0xFFFFFFFFu - (unsigned)vcol);  // first-idx ties
#pragma unroll
          for (int off = 1; off < 16; off <<= 1) {
            unsigned long long o = __shfl_xor(pk, off);
            pk = (o > pk) ? o : pk;
          }
          if (l16 == 0) atomicMax(&pmax[m], pk);
        }
      }
    }
    __syncthreads();

    // ---- phase 2b: sum exp(x - M) via ds_add_f32 ----
    for (int q = 0; q < 2; ++q) {
#pragma unroll
      for (int r = 0; r < 2; ++r) {
#pragma unroll
        for (int v = 0; v < 8; ++v) {
          int m = r * 16 + v + 8 * half;
          unsigned int kk = (unsigned)(pmax[m] >> 32);
          unsigned int uu = (kk & 0x80000000u) ? (kk & 0x7fffffffu) : ~kk;
          float M = __uint_as_float(uu);
          float e = __expf(st[q][r][v] - M);
#pragma unroll
          for (int off = 1; off < 16; off <<= 1) e += __shfl_xor(e, off);
          if (l16 == 0) atomicAdd(&sumexp[m], e);
        }
      }
    }
    __syncthreads();

    // ---- phase 2c: per-row greedy decision ----
    if (tid < 32) {
      unsigned long long pk = pmax[tid];
      int pos = (int)(0xFFFFFFFFu - (unsigned)(pk & 0xFFFFFFFFull));
      float lp = -__logf(sumexp[tid]);  // logit[argmax] - logsumexp
      int nb = (pos != 0);
      if (nb) score[tid] += lp;
      int tk = nb ? pos : lastS[tid];
      lastS[tid] = tk;
      tokS[tid] = tk;
      nbS[tid] = nb;
      out[tid * TT + t] = (float)(nb ? pos : 0);
    }
    __syncthreads();

    // ---- phase 3: gather x = E[tok] into zxb (bf16, reuses z buffer) ----
    for (int i = tid; i < BB * (DD / 4); i += BDIM) {
      int row = i / (DD / 4);
      int c4 = i % (DD / 4);
      int tk = tokS[row];
      float4 fv = ((const float4*)(E + (size_t)tk * DD))[c4];
      v4bf w; w[0] = (__bf16)fv.x; w[1] = (__bf16)fv.y; w[2] = (__bf16)fv.z; w[3] = (__bf16)fv.w;
      *(v4bf*)(zxb + (size_t)i * 4) = w;
    }
    __syncthreads();

    // ---- phase 4: masked LSTM step, merged h into other bf16 buffer ----
    lstm_phase(zxb, hc, hnx, Wih, Whh, bl, creg, nbS, false, wave, l16, half);
    __syncthreads();
  }

  if (tid < 32) out[BB * TT + tid] = score[tid];
  if (tid == 0) {
    float s = 0.0f;
    for (int i = 0; i < 32; ++i) s += __expf(score[i]);
    out[BB * TT + BB] = s / 32.0f;
  }
}

extern "C" void kernel_launch(void* const* d_in, const int* in_sizes, int n_in,
                              void* d_out, int out_size, void* d_ws,
                              size_t ws_size, hipStream_t stream) {
  const float* tn   = (const float*)d_in[0];
  const float* E    = (const float*)d_in[1];
  const float* Wih  = (const float*)d_in[2];
  const float* Whh  = (const float*)d_in[3];
  const float* bl   = (const float*)d_in[4];
  const float* Wtn  = (const float*)d_in[5];
  const float* Wpn  = (const float*)d_in[6];
  const float* bj   = (const float*)d_in[7];
  const float* Wout = (const float*)d_in[8];
  const float* bo   = (const float*)d_in[9];

  size_t smem = 163840 + 1024;  // 4x 40KB bf16 activation buffers + scratch
  (void)hipFuncSetAttribute((const void*)rnnt_decode,
                            hipFuncAttributeMaxDynamicSharedMemorySize,
                            (int)smem);
  rnnt_decode<<<1, BDIM, smem, stream>>>(tn, E, Wih, Whh, bl, Wtn, Wpn, bj,
                                         Wout, bo, (float*)d_out);
}
